// CNN_90864328114608
// MI455X (gfx1250) — compile-verified, matrix operand order checked by reading, not run on hardware
//
#include <hip/hip_runtime.h>
#include <hip/hip_bf16.h>

// ---------------------------------------------------------------------------
// PCNN relation-extraction forward pass for gfx1250 (MI455X).
// Conv1d(310ch -> 230f, k=3, edge pad) cast as f16 WMMA GEMM, fused with
// embedding gather (LDS staging), piecewise max-pool, bag segment-sum.
//
// Each wave owns TWO M-tiles (mt = wave, wave+8) processed together: per
// K-step, 2 A-fragments (weights, L2) x 8 B-fragments (LDS) -> 16 WMMAs.
// B-fragments are consumed immediately after load (no cross-loop invariance,
// so no LICM spill-to-scratch), and each B load feeds 2 WMMAs.
// ---------------------------------------------------------------------------

#define MAX_LEN   120
#define NF        230
#define MP        240     // pool stride (NF padded to 16)
#define WPAD      256     // weight rows padded so every wave has 2 tiles
#define IN_CH     310
#define CP        320     // IN_CH padded to 32
#define NREL      53
#define NSENT     2048
#define NBAGS     256
#define LROWS     130     // 120 + edge pads + tap-shift slack

typedef __attribute__((ext_vector_type(16))) _Float16 v16h;
typedef __attribute__((ext_vector_type(4)))  _Float16 v4h;
typedef __attribute__((ext_vector_type(8)))  float    v8f;
typedef __attribute__((ext_vector_type(4)))  unsigned u32x4;

union Frag { v16h h; u32x4 q[2]; };

__device__ __forceinline__ unsigned enc_f32(float v) {
    unsigned u = __float_as_uint(v);
    return (u & 0x80000000u) ? ~u : (u | 0x80000000u);
}

// ---------------------------------------------------------------------------
// Kernel 0: zero the bag accumulators + convert/pad conv weights to f16.
// wH layout: [k:3][m:256][c:320], zero outside (m<230, c<310).
// ---------------------------------------------------------------------------
__global__ __launch_bounds__(256)
void prep_kernel(const float* __restrict__ conv_w,
                 float* __restrict__ sums, float* __restrict__ cnt,
                 _Float16* __restrict__ wH) {
    int idx = blockIdx.x * 256 + threadIdx.x;
    if (idx < 3 * WPAD * CP) {
        int k = idx / (WPAD * CP);
        int rem = idx % (WPAD * CP);
        int m = rem / CP, c = rem % CP;
        _Float16 v = (_Float16)0.0f;
        if (m < NF && c < IN_CH)
            v = (_Float16)conv_w[(m * IN_CH + c) * 3 + k];   // conv_w [NF][IN_CH][3]
        wH[idx] = v;
    }
    if (idx < NBAGS * 3 * NF) sums[idx] = 0.0f;
    if (idx < NBAGS)          cnt[idx]  = 0.0f;
}

// ---------------------------------------------------------------------------
// Kernel 1: one sentence per workgroup; 8 wave32; dual M-tile per wave.
// ---------------------------------------------------------------------------
__global__ __launch_bounds__(256)
void pcnn_kernel(const int*   __restrict__ sentences,
                 const int*   __restrict__ pos1,
                 const int*   __restrict__ pos2,
                 const float* __restrict__ masks,     // [N][3][120]
                 const int*   __restrict__ bag_ids,
                 const float* __restrict__ word_emb,  // [VOCAB][300]
                 const float* __restrict__ pf1_emb,   // [240][5]
                 const float* __restrict__ pf2_emb,   // [240][5]
                 const float* __restrict__ conv_b,    // [230]
                 const _Float16* __restrict__ wH,     // [3][256][320] f16
                 float* __restrict__ sums,            // [256][690]
                 float* __restrict__ cnt) {           // [256]
    __shared__ _Float16 sX[LROWS * CP];       // 83,200 B : padded xT[row][chan]
    __shared__ unsigned sPool[3 * MP];        //  2,880 B : piecewise max keys
    __shared__ int      sPiece[128];          //    512 B : piece id per column

    const int s   = blockIdx.x;
    const int tid = threadIdx.x;
    const int* sent = sentences + s * MAX_LEN;
    const int* p1   = pos1      + s * MAX_LEN;
    const int* p2   = pos2      + s * MAX_LEN;

    // ---- Phase 1: gather embeddings into LDS (row j = edge-padded column j-1)
    for (int t = tid; t < LROWS * 75; t += 256) {      // 300 floats = 75 float4
        int j = t / 75, q = t % 75;
        int l = j - 1; l = l < 0 ? 0 : (l > MAX_LEN - 1 ? MAX_LEN - 1 : l);
        const float4 w4 = *(const float4*)(word_emb + sent[l] * 300 + q * 4);
        v4h h; h[0] = (_Float16)w4.x; h[1] = (_Float16)w4.y;
               h[2] = (_Float16)w4.z; h[3] = (_Float16)w4.w;
        *(v4h*)(&sX[j * CP + q * 4]) = h;
    }
    for (int t = tid; t < LROWS * 20; t += 256) {      // pf1/pf2 + zero pad
        int j = t / 20, c2 = t % 20;
        int l = j - 1; l = l < 0 ? 0 : (l > MAX_LEN - 1 ? MAX_LEN - 1 : l);
        float v = 0.0f;
        if (c2 < 5)       v = pf1_emb[p1[l] * 5 + c2];
        else if (c2 < 10) v = pf2_emb[p2[l] * 5 + (c2 - 5)];
        sX[j * CP + 300 + c2] = (_Float16)v;
    }
    for (int i = tid; i < 3 * MP; i += 256) sPool[i] = 0u;   // key 0 ~ -inf
    for (int i = tid; i < 128; i += 256) {
        int pc = 3;                                           // 3 = invalid column
        if (i < MAX_LEN) {
            float m1 = masks[(s * 3 + 1) * MAX_LEN + i];
            float m2 = masks[(s * 3 + 2) * MAX_LEN + i];
            pc = (m1 > 0.5f ? 1 : 0) + (m2 > 0.5f ? 2 : 0);
        }
        sPiece[i] = pc;
    }
    __syncthreads();

    // ---- Phase 2: WMMA GEMM, dual M-tile per wave
    const int wave = tid >> 5;
    const int lane = tid & 31;
    const int ln   = lane & 15;
    const int hf   = lane >> 4;                // half-wave selector
    const int mt0  = wave;                     // M-tile 0
    const int mt1  = wave + 8;                 // M-tile 1 (15 = pad tile)

    v8f acc[2][8];
    #pragma unroll
    for (int t = 0; t < 2; ++t)
        #pragma unroll
        for (int nt = 0; nt < 8; ++nt)
            #pragma unroll
            for (int r = 0; r < 8; ++r) acc[t][nt][r] = 0.0f;

    #pragma unroll 1
    for (int k = 0; k < 3; ++k) {
        const _Float16* wk0 = wH + k * (WPAD * CP) + (mt0 * 16 + ln) * CP;
        const _Float16* wk1 = wH + k * (WPAD * CP) + (mt1 * 16 + ln) * CP;
        const _Float16* xk  = &sX[(ln + k) * CP];
        #pragma unroll 1
        for (int kk = 0; kk < 10; ++kk) {
            const int kb = kk * 32 + hf * 8;   // K{0..7}/{8..15} per half-wave
            Frag a0, a1;
            a0.q[0] = *(const u32x4*)(wk0 + kb);
            a0.q[1] = *(const u32x4*)(wk0 + kb + 16);
            a1.q[0] = *(const u32x4*)(wk1 + kb);
            a1.q[1] = *(const u32x4*)(wk1 + kb + 16);
            #pragma unroll
            for (int nt = 0; nt < 8; ++nt) {
                Frag b;
                const _Float16* bp = xk + nt * (16 * CP) + kb;
                b.q[0] = *(const u32x4*)(bp);
                b.q[1] = *(const u32x4*)(bp + 16);
                acc[0][nt] = __builtin_amdgcn_wmma_f32_16x16x32_f16(
                               false, a0.h, false, b.h, (short)0, acc[0][nt],
                               false, false);
                acc[1][nt] = __builtin_amdgcn_wmma_f32_16x16x32_f16(
                               false, a1.h, false, b.h, (short)0, acc[1][nt],
                               false, false);
            }
        }
    }

    // ---- Phase 3: fused piecewise max-pool
    // C layout: column = nt*16 + ln, row m = mt*16 + hf*8 + r
    #pragma unroll
    for (int t = 0; t < 2; ++t) {
        const int mtv = (t == 0) ? mt0 : mt1;
        #pragma unroll
        for (int nt = 0; nt < 8; ++nt) {
            const int pc = sPiece[nt * 16 + ln];
            if (pc < 3) {
                #pragma unroll
                for (int r = 0; r < 8; ++r) {
                    int m = mtv * 16 + hf * 8 + r;
                    if (m < NF)   // also discards pad tile mt=15 entirely
                        atomicMax(&sPool[pc * MP + m], enc_f32(acc[t][nt][r]));
                }
            }
        }
    }
    __syncthreads();

    // ---- Phase 4: bias + ReLU + bag segment-sum (global f32 atomics)
    const int bag = bag_ids[s];
    for (int i = tid; i < 3 * NF; i += 256) {
        int p = i / NF, nf = i % NF;
        unsigned key = sPool[p * MP + nf];
        float v;
        if (key == 0u) v = -3.0e38f;                      // empty piece
        else {
            unsigned u = (key & 0x80000000u) ? (key & 0x7FFFFFFFu) : ~key;
            v = __uint_as_float(u);
        }
        v = v + conv_b[nf];
        v = v > 0.0f ? v : 0.0f;                          // relu
        atomicAdd(&sums[bag * (3 * NF) + i], v);
    }
    if (tid == 0) atomicAdd(&cnt[bag], 1.0f);
}

// ---------------------------------------------------------------------------
// Kernel 2: per-bag mean -> dense (53x690) -> softmax.  One bag per block.
// ---------------------------------------------------------------------------
__global__ __launch_bounds__(64)
void finish_kernel(const float* __restrict__ sums, const float* __restrict__ cnt,
                   const float* __restrict__ dense_w, const float* __restrict__ dense_b,
                   float* __restrict__ out) {
    __shared__ float repre[3 * NF];
    __shared__ float red[64];
    const int b   = blockIdx.x;
    const int tid = threadIdx.x;
    const float inv = 1.0f / fmaxf(cnt[b], 1.0f);
    for (int i = tid; i < 3 * NF; i += 64) repre[i] = sums[b * (3 * NF) + i] * inv;
    __syncthreads();

    float lg = -3.0e38f;
    if (tid < NREL) {
        float acc = dense_b[tid];
        const float* wrow = dense_w + tid * (3 * NF);
        for (int i = 0; i < 3 * NF; ++i) acc += wrow[i] * repre[i];
        lg = acc;
    }
    red[tid] = lg;
    __syncthreads();
    for (int ofs = 32; ofs > 0; ofs >>= 1) {
        if (tid < ofs) red[tid] = fmaxf(red[tid], red[tid + ofs]);
        __syncthreads();
    }
    const float mx = red[0];
    __syncthreads();
    const float e = (tid < NREL) ? __expf(lg - mx) : 0.0f;
    red[tid] = e;
    __syncthreads();
    for (int ofs = 32; ofs > 0; ofs >>= 1) {
        if (tid < ofs) red[tid] += red[tid + ofs];
        __syncthreads();
    }
    if (tid < NREL) out[b * NREL + tid] = e / red[0];
}

// ---------------------------------------------------------------------------
extern "C" void kernel_launch(void* const* d_in, const int* in_sizes, int n_in,
                              void* d_out, int out_size, void* d_ws, size_t ws_size,
                              hipStream_t stream) {
    const int*   sentences = (const int*)  d_in[0];
    const int*   pos1      = (const int*)  d_in[1];
    const int*   pos2      = (const int*)  d_in[2];
    const float* masks     = (const float*)d_in[3];
    const int*   bag_ids   = (const int*)  d_in[4];
    const float* word_emb  = (const float*)d_in[5];
    const float* pf1_emb   = (const float*)d_in[6];
    const float* pf2_emb   = (const float*)d_in[7];
    const float* conv_w    = (const float*)d_in[8];
    const float* conv_b    = (const float*)d_in[9];
    const float* dense_w   = (const float*)d_in[10];
    const float* dense_b   = (const float*)d_in[11];
    float* out = (float*)d_out;

    // workspace: sums[256*690] f32 | cnt[256] f32 | wH[3*256*320] f16
    float*    sums = (float*)d_ws;
    float*    cnt  = sums + NBAGS * 3 * NF;
    _Float16* wH   = (_Float16*)(cnt + NBAGS);

    prep_kernel<<<(3 * WPAD * CP + 255) / 256, 256, 0, stream>>>(conv_w, sums, cnt, wH);
    pcnn_kernel<<<NSENT, 256, 0, stream>>>(sentences, pos1, pos2, masks, bag_ids,
                                           word_emb, pf1_emb, pf2_emb, conv_b,
                                           wH, sums, cnt);
    finish_kernel<<<NBAGS, 64, 0, stream>>>(sums, cnt, dense_w, dense_b, out);
}
